// Hgrn2ViT_53884659695721
// MI455X (gfx1250) — matrix operand order repaired and checked by reading, not compiled, not run on hardware
//
#include <hip/hip_runtime.h>
#include <math.h>

typedef __attribute__((ext_vector_type(16))) __bf16 v16bf;
typedef __attribute__((ext_vector_type(8)))  float  v8f;

#define BATCH   64
#define SEQ     197
#define NPATCH  196
#define ROWS    (BATCH*SEQ)     /* 12608 */
#define RROWS   12672           /* ROWS padded to 99*128 so A-tile loads need no guard */
#define PROWS   (BATCH*NPATCH)  /* 12544 = 98*128, already exact */
#define DIMV    768
#define FD      128
#define MLPD    3072
#define PD      3840
#define NDEPTH  12
#define NCLASS  1000

static __device__ __forceinline__ unsigned short f2bf(float f) {
  unsigned u = __float_as_uint(f);
  u += 0x7FFFu + ((u >> 16) & 1u);          // round-to-nearest-even
  return (unsigned short)(u >> 16);
}
static __device__ __forceinline__ float gelu_exact(float x) {
  return 0.5f * x * (1.0f + erff(x * 0.70710678118654752440f));
}

// ---------------------------------------------------------------------------
// WMMA GEMM: C[M,N] = A[M,K](bf16) * Bt[N,K](bf16)^T  + epilogue
// Block tile 128x128, 8 waves (wave32), wave tile 32x64 = 2x4 WMMA 16x16x32.
// Register-staged double buffering: next global tile is in flight while the
// current LDS tile feeds the WMMAs. A rows are padded -> branch-free staging.
// ---------------------------------------------------------------------------
#define BM 128
#define BN 128
#define BK 32
#define LSTR 40   // LDS row stride in bf16 elems (80B: 16B-aligned, conflict-free)

enum { EPI_F32 = 0, EPI_BIAS_GELU_BF16 = 1, EPI_BIAS_RESID = 2, EPI_RESID = 3, EPI_PATCH = 4 };

template <int EPI>
__global__ __launch_bounds__(256) void k_gemm(
    const unsigned short* __restrict__ A, const unsigned short* __restrict__ Bt,
    void* __restrict__ Cv, int M, int N, int K,
    const float* __restrict__ bias, const float* __restrict__ resid,
    const float* __restrict__ pos)
{
  __shared__ __align__(16) unsigned short As[BM * LSTR];
  __shared__ __align__(16) unsigned short Bs[BN * LSTR];

  const int tid   = threadIdx.x;
  const int lane  = tid & 31, wave = tid >> 5;
  const int wm    = (wave & 3) * 32;   // 4 wave-rows  -> 128
  const int wn    = (wave >> 2) * 64;  // 2 wave-cols  -> 128
  const int lrow  = lane & 15, lhalf = lane >> 4;
  const int bm0   = blockIdx.x * BM;
  const int bn0   = blockIdx.y * BN;

  // Per-thread staging coords: chunks tid and tid+256 -> rows r0, r0+64, same col.
  const int r0 = tid >> 2;
  const int c0 = (tid & 3) * 8;
  const unsigned short* Ag0 = A  + (size_t)(bm0 + r0)      * K + c0;
  const unsigned short* Ag1 = A  + (size_t)(bm0 + r0 + 64) * K + c0;
  const unsigned short* Bg0 = Bt + (size_t)(bn0 + r0)      * K + c0;
  const unsigned short* Bg1 = Bt + (size_t)(bn0 + r0 + 64) * K + c0;
  unsigned short* Al0 = &As[r0 * LSTR + c0];
  unsigned short* Al1 = &As[(r0 + 64) * LSTR + c0];
  unsigned short* Bl0 = &Bs[r0 * LSTR + c0];
  unsigned short* Bl1 = &Bs[(r0 + 64) * LSTR + c0];

  const v8f vz = {0.f,0.f,0.f,0.f,0.f,0.f,0.f,0.f};
  v8f acc[2][4];
#pragma unroll
  for (int mt = 0; mt < 2; ++mt)
#pragma unroll
    for (int nt = 0; nt < 4; ++nt) acc[mt][nt] = vz;

  union Frag { uint4 u[2]; v16bf v; };
  auto compute = [&]() {
    Frag fa[2], fb[4];
    // A fragment (ISA 16-bit A 16x32): e0..7 -> K=lhalf*8+0..7 ; e8..15 -> K=16+lhalf*8+0..7
#pragma unroll
    for (int mt = 0; mt < 2; ++mt) {
      int m = wm + mt * 16 + lrow;
      fa[mt].u[0] = *(const uint4*)(&As[m * LSTR + lhalf * 8]);
      fa[mt].u[1] = *(const uint4*)(&As[m * LSTR + 16 + lhalf * 8]);
    }
    // B fragment (ISA 16-bit B 32x16): e0..7 -> K=lhalf*8+0..7 ; e8..15 -> K=16+lhalf*8+0..7
#pragma unroll
    for (int nt = 0; nt < 4; ++nt) {
      int n = wn + nt * 16 + lrow;
      fb[nt].u[0] = *(const uint4*)(&Bs[n * LSTR + lhalf * 8]);
      fb[nt].u[1] = *(const uint4*)(&Bs[n * LSTR + 16 + lhalf * 8]);
    }
#pragma unroll
    for (int mt = 0; mt < 2; ++mt)
#pragma unroll
      for (int nt = 0; nt < 4; ++nt)
        acc[mt][nt] = __builtin_amdgcn_wmma_f32_16x16x32_bf16(
            false, fa[mt].v, false, fb[nt].v, (short)0, acc[mt][nt], false, false);
  };

  // Prologue: stage tile k0=0 straight into LDS (A padded -> no guards anywhere).
  {
    uint4 a0 = *(const uint4*)Ag0, a1 = *(const uint4*)Ag1;
    uint4 b0 = *(const uint4*)Bg0, b1 = *(const uint4*)Bg1;
    *(uint4*)Al0 = a0; *(uint4*)Al1 = a1;
    *(uint4*)Bl0 = b0; *(uint4*)Bl1 = b1;
  }
  __syncthreads();

  int k0 = 0;
  for (; k0 + BK < K; k0 += BK) {
    // issue next tile's global loads (latency hidden behind WMMAs)
    uint4 a0 = *(const uint4*)(Ag0 + k0 + BK);
    uint4 a1 = *(const uint4*)(Ag1 + k0 + BK);
    uint4 b0 = *(const uint4*)(Bg0 + k0 + BK);
    uint4 b1 = *(const uint4*)(Bg1 + k0 + BK);
    compute();
    __syncthreads();
    *(uint4*)Al0 = a0; *(uint4*)Al1 = a1;
    *(uint4*)Bl0 = b0; *(uint4*)Bl1 = b1;
    __syncthreads();
  }
  compute();  // last tile

  // ---- epilogue: D layout -> row m = base + lhalf*8 + v, col n = base + lrow ----
#pragma unroll
  for (int mt = 0; mt < 2; ++mt) {
#pragma unroll
    for (int nt = 0; nt < 4; ++nt) {
      union { v8f v; float f[8]; } u; u.v = acc[mt][nt];
      int gc = bn0 + wn + nt * 16 + lrow;
#pragma unroll
      for (int vv = 0; vv < 8; ++vv) {
        int gr = bm0 + wm + mt * 16 + lhalf * 8 + vv;
        if (gr >= M) continue;
        float a = u.f[vv];
        if (EPI == EPI_F32) {
          ((float*)Cv)[(size_t)gr * N + gc] = a;
        } else if (EPI == EPI_BIAS_GELU_BF16) {
          ((unsigned short*)Cv)[(size_t)gr * N + gc] = f2bf(gelu_exact(a + bias[gc]));
        } else if (EPI == EPI_BIAS_RESID) {
          ((float*)Cv)[(size_t)gr * N + gc] = a + bias[gc] + resid[(size_t)gr * N + gc];
        } else if (EPI == EPI_RESID) {
          ((float*)Cv)[(size_t)gr * N + gc] = a + resid[(size_t)gr * N + gc];
        } else { // EPI_PATCH: remap 196-row batches into 197-row x, add pos_emb
          int b = gr / NPATCH, l = gr % NPATCH;
          ((float*)Cv)[((size_t)b * SEQ + l) * N + gc] = a + pos[(size_t)l * N + gc];
        }
      }
    }
  }
}

// ---------------------------------------------------------------------------
// Weight convert + transpose: W[K,N] fp32 -> Wt[N,K] bf16
// ---------------------------------------------------------------------------
__global__ void k_cvtT(const float* __restrict__ W, unsigned short* __restrict__ Wt,
                       int K, int N) {
  int id = blockIdx.x * 256 + threadIdx.x;
  if (id >= K * N) return;
  int n = id / K, k = id - n * K;
  Wt[id] = f2bf(W[(size_t)k * N + n]);
}

// lb = cumsum(softmax(lower_bounds, axis=0)) - row0   (DEPTH x FD)
__global__ void k_lb(const float* __restrict__ lbnd, float* __restrict__ lb) {
  int f = threadIdx.x;
  if (f >= FD) return;
  float v[NDEPTH]; float mx = -1e30f;
  for (int d = 0; d < NDEPTH; ++d) { v[d] = lbnd[d * FD + f]; mx = fmaxf(mx, v[d]); }
  float s = 0.f;
  for (int d = 0; d < NDEPTH; ++d) { v[d] = expf(v[d] - mx); s += v[d]; }
  float inv = 1.f / s, cum = 0.f, first = 0.f;
  for (int d = 0; d < NDEPTH; ++d) {
    cum += v[d] * inv;
    if (d == 0) first = cum;
    lb[d * FD + f] = cum - first;
  }
}

// Build layernormed 5C shifted patches -> bf16 A matrix (PROWS x 3840)
__global__ __launch_bounds__(256) void k_patches(
    const float* __restrict__ img, const float* __restrict__ g,
    const float* __restrict__ bta, unsigned short* __restrict__ Pn) {
  const int p = blockIdx.x;            // 0..PROWS-1
  const int b = p / NPATCH, pp = p % NPATCH;
  const int ph = pp / 14, pw = pp % 14;
  const int dh[5] = {0, 0, 0, 1, -1};
  const int dw[5] = {0, 1, -1, 0, 0};
  const int tid = threadIdx.x;
  float vals[15]; float s = 0.f, s2 = 0.f;
#pragma unroll
  for (int i = 0; i < 15; ++i) {
    int d = tid + 256 * i;              // (py,px,c5) flat
    int pix = d / 15, c5 = d % 15;
    int py = pix >> 4, px = pix & 15;
    int grp = c5 / 3, c0 = c5 % 3;
    int y = ph * 16 + py - dh[grp];
    int x = pw * 16 + px - dw[grp];
    float v = 0.f;
    if (y >= 0 && y < 224 && x >= 0 && x < 224)
      v = img[((size_t)(b * 3 + c0) * 224 + y) * 224 + x];
    vals[i] = v; s += v; s2 += v * v;
  }
  __shared__ float r1[256], r2[256];
  r1[tid] = s; r2[tid] = s2; __syncthreads();
  for (int o = 128; o > 0; o >>= 1) {
    if (tid < o) { r1[tid] += r1[tid + o]; r2[tid] += r2[tid + o]; }
    __syncthreads();
  }
  float mean = r1[0] / (float)PD;
  float var  = r2[0] / (float)PD - mean * mean;
  float inv  = rsqrtf(var + 1e-5f);
#pragma unroll
  for (int i = 0; i < 15; ++i) {
    int d = tid + 256 * i;
    Pn[(size_t)p * PD + d] = f2bf((vals[i] - mean) * inv * g[d] + bta[d]);
  }
}

// x[:,196,:] = cls + pos[196]
__global__ void k_cls_pos(const float* __restrict__ cls, const float* __restrict__ pos,
                          float* __restrict__ x) {
  int n = blockIdx.x * 256 + threadIdx.x;  // 64*768
  int b = n / DIMV, d = n % DIMV;
  x[((size_t)b * SEQ + NPATCH) * DIMV + d] = cls[d] + pos[(size_t)NPATCH * DIMV + d];
}

// rmsnorm each row of x (ROWS x 768) -> bf16
__global__ __launch_bounds__(256) void k_rmsnorm(const float* __restrict__ x,
                                                 unsigned short* __restrict__ xn) {
  int r = blockIdx.x, tid = threadIdx.x;
  const float* xr = x + (size_t)r * DIMV;
  float v[3]; float ss = 0.f;
#pragma unroll
  for (int i = 0; i < 3; ++i) { v[i] = xr[tid + 256 * i]; ss += v[i] * v[i]; }
  __shared__ float red[256];
  red[tid] = ss; __syncthreads();
  for (int o = 128; o > 0; o >>= 1) {
    if (tid < o) red[tid] += red[tid + o];
    __syncthreads();
  }
  float inv = rsqrtf(red[0] / (float)DIMV + 1e-6f);
  unsigned short* on = xn + (size_t)r * DIMV;
#pragma unroll
  for (int i = 0; i < 3; ++i) on[tid + 256 * i] = f2bf(v[i] * inv);
}

// GLA scan: one lane per (batch, head); fuses gelu(q), gating, k, log-gate,
// the 2x2-state recurrence, and tanh -> bf16 output (ROWS x 128).
__global__ __launch_bounds__(256) void k_scan(const float* __restrict__ qfv,
                                              const float* __restrict__ lbrow,
                                              unsigned short* __restrict__ O) {
  int t0 = blockIdx.x * 256 + threadIdx.x;  // 0..4095
  int b = t0 >> 6, h = t0 & 63;
  int c = h * 2;
  float lb0 = lbrow[c], lb1 = lbrow[c + 1];
  float S00 = 0.f, S01 = 0.f, S10 = 0.f, S11 = 0.f;
  const float scale = 0.70710678118654752440f;  // ER^-0.5
  for (int t = 0; t < SEQ; ++t) {
    const float* row = qfv + (size_t)(b * SEQ + t) * 384;
    float q0 = gelu_exact(row[c]);
    float q1 = gelu_exact(row[c + 1]);
    float f0 = row[128 + c], f1 = row[128 + c + 1];
    float v0 = row[256 + c], v1 = row[256 + c + 1];
    float g0 = lb0 + (1.f - lb0) * (1.f / (1.f + expf(-f0)));
    float g1 = lb1 + (1.f - lb1) * (1.f / (1.f + expf(-f1)));
    float k0 = 1.f - g0, k1 = 1.f - g1;
    float e0 = g0 + 1e-6f, e1 = g1 + 1e-6f;   // exp(log(g+1e-6))
    S00 = e0 * S00 + k0 * v0;  S01 = e0 * S01 + k0 * v1;
    S10 = e1 * S10 + k1 * v0;  S11 = e1 * S11 + k1 * v1;
    float o0 = scale * (q0 * S00 + q1 * S10);
    float o1 = scale * (q0 * S01 + q1 * S11);
    unsigned short* orow = O + (size_t)(b * SEQ + t) * FD + c;
    orow[0] = f2bf(tanhf(o0));
    orow[1] = f2bf(tanhf(o1));
  }
}

// Final per-batch: rmsnorm(last token) then layernorm(g,b) -> hx (64 x 768)
__global__ __launch_bounds__(256) void k_final(const float* __restrict__ x,
                                               const float* __restrict__ g,
                                               const float* __restrict__ be,
                                               float* __restrict__ hx) {
  int b = blockIdx.x, tid = threadIdx.x;
  const float* xr = x + ((size_t)b * SEQ + (SEQ - 1)) * DIMV;
  float v[3]; float ss = 0.f;
#pragma unroll
  for (int i = 0; i < 3; ++i) { v[i] = xr[tid + 256 * i]; ss += v[i] * v[i]; }
  __shared__ float r1[256], r2[256];
  r1[tid] = ss; __syncthreads();
  for (int o = 128; o > 0; o >>= 1) {
    if (tid < o) r1[tid] += r1[tid + o];
    __syncthreads();
  }
  float inv = rsqrtf(r1[0] / (float)DIMV + 1e-6f);
  __syncthreads();
  float s = 0.f, q = 0.f;
#pragma unroll
  for (int i = 0; i < 3; ++i) { v[i] *= inv; s += v[i]; q += v[i] * v[i]; }
  r1[tid] = s; r2[tid] = q; __syncthreads();
  for (int o = 128; o > 0; o >>= 1) {
    if (tid < o) { r1[tid] += r1[tid + o]; r2[tid] += r2[tid + o]; }
    __syncthreads();
  }
  float mean = r1[0] / (float)DIMV;
  float var  = r2[0] / (float)DIMV - mean * mean;
  float inv2 = rsqrtf(var + 1e-5f);
#pragma unroll
  for (int i = 0; i < 3; ++i) {
    int d = tid + 256 * i;
    hx[(size_t)b * DIMV + d] = (v[i] - mean) * inv2 * g[d] + be[d];
  }
}

// Classifier head: out[b,n] = hx[b,:] . W_head[:,n] + b_head[n]  (tiny, fp32)
__global__ __launch_bounds__(256) void k_head(const float* __restrict__ hx,
                                              const float* __restrict__ Wh,
                                              const float* __restrict__ bh,
                                              float* __restrict__ out) {
  int b = blockIdx.x;
  __shared__ float row[DIMV];
  for (int i = threadIdx.x; i < DIMV; i += 256) row[i] = hx[(size_t)b * DIMV + i];
  __syncthreads();
  for (int n = threadIdx.x; n < NCLASS; n += 256) {
    float acc = bh[n];
    for (int k = 0; k < DIMV; ++k) acc += row[k] * Wh[(size_t)k * NCLASS + n];
    out[(size_t)b * NCLASS + n] = acc;
  }
}

// ---------------------------------------------------------------------------
extern "C" void kernel_launch(void* const* d_in, const int* in_sizes, int n_in,
                              void* d_out, int out_size, void* d_ws, size_t ws_size,
                              hipStream_t stream) {
  (void)in_sizes; (void)n_in; (void)out_size; (void)ws_size;
  const float* img   = (const float*)d_in[0];
  const float* lnp_g = (const float*)d_in[1];
  const float* lnp_b = (const float*)d_in[2];
  const float* Wp    = (const float*)d_in[3];
  const float* pos   = (const float*)d_in[4];
  const float* cls   = (const float*)d_in[5];
  const float* lbnd  = (const float*)d_in[6];
  const float* Wq    = (const float*)d_in[7];
  const float* Wf    = (const float*)d_in[8];
  const float* Wi    = (const float*)d_in[9];
  const float* Wo    = (const float*)d_in[10];
  const float* W1    = (const float*)d_in[11];
  const float* b1    = (const float*)d_in[12];
  const float* W2    = (const float*)d_in[13];
  const float* b2    = (const float*)d_in[14];
  const float* lnh_g = (const float*)d_in[15];
  const float* lnh_b = (const float*)d_in[16];
  const float* Wh    = (const float*)d_in[17];
  const float* bh    = (const float*)d_in[18];

  char* ws = (char*)d_ws;
  size_t off = 0;
  auto alloc = [&](size_t bytes) -> char* {
    char* p = ws + off;
    off += (bytes + 255) & ~(size_t)255;
    return p;
  };
  float*          lb    = (float*)alloc((size_t)NDEPTH * FD * 4);
  unsigned short* WpT   = (unsigned short*)alloc((size_t)DIMV * PD * 2);
  unsigned short* WqfvT = (unsigned short*)alloc((size_t)NDEPTH * 384 * DIMV * 2);
  unsigned short* WoT   = (unsigned short*)alloc((size_t)NDEPTH * DIMV * FD * 2);
  unsigned short* W1T   = (unsigned short*)alloc((size_t)NDEPTH * MLPD * DIMV * 2);
  unsigned short* W2T   = (unsigned short*)alloc((size_t)NDEPTH * DIMV * MLPD * 2);
  unsigned short* Pn    = (unsigned short*)alloc((size_t)PROWS * PD * 2);
  float*          x     = (float*)alloc((size_t)ROWS * DIMV * 4);
  unsigned short* xn    = (unsigned short*)alloc((size_t)RROWS * DIMV * 2);  // padded A
  float*          qfv   = (float*)alloc((size_t)ROWS * 384 * 4);
  unsigned short* Obuf  = (unsigned short*)alloc((size_t)RROWS * FD * 2);    // padded A
  unsigned short* hbuf  = (unsigned short*)alloc((size_t)RROWS * MLPD * 2);  // padded A
  float*          hx    = (float*)alloc((size_t)BATCH * DIMV * 4);

  auto cvt = [&](const float* W, unsigned short* Wt, int K, int N) {
    k_cvtT<<<(K * N + 255) / 256, 256, 0, stream>>>(W, Wt, K, N);
  };

  // gate schedule + weight conversion (fp32 -> transposed bf16)
  k_lb<<<1, 128, 0, stream>>>(lbnd, lb);
  cvt(Wp, WpT, PD, DIMV);
  for (int i = 0; i < NDEPTH; ++i) {
    unsigned short* wq = WqfvT + (size_t)i * 384 * DIMV;
    cvt(Wq + (size_t)i * DIMV * FD, wq,                          DIMV, FD);
    cvt(Wf + (size_t)i * DIMV * FD, wq + (size_t)FD * DIMV,      DIMV, FD);
    cvt(Wi + (size_t)i * DIMV * FD, wq + (size_t)2 * FD * DIMV,  DIMV, FD);
    cvt(Wo + (size_t)i * FD * DIMV,   WoT + (size_t)i * DIMV * FD,   FD,   DIMV);
    cvt(W1 + (size_t)i * DIMV * MLPD, W1T + (size_t)i * MLPD * DIMV, DIMV, MLPD);
    cvt(W2 + (size_t)i * MLPD * DIMV, W2T + (size_t)i * DIMV * MLPD, MLPD, DIMV);
  }

  // patch embed
  k_patches<<<PROWS, 256, 0, stream>>>(img, lnp_g, lnp_b, Pn);
  k_gemm<EPI_PATCH><<<dim3(PROWS / BM, DIMV / BN), 256, 0, stream>>>(
      Pn, WpT, x, PROWS, DIMV, PD, nullptr, nullptr, pos);
  k_cls_pos<<<(BATCH * DIMV) / 256, 256, 0, stream>>>(cls, pos, x);

  const int gm = RROWS / BM;  // 99
  for (int i = 0; i < NDEPTH; ++i) {
    k_rmsnorm<<<ROWS, 256, 0, stream>>>(x, xn);
    k_gemm<EPI_F32><<<dim3(gm, 384 / BN), 256, 0, stream>>>(
        xn, WqfvT + (size_t)i * 384 * DIMV, qfv, ROWS, 384, DIMV,
        nullptr, nullptr, nullptr);
    k_scan<<<(BATCH * 64) / 256, 256, 0, stream>>>(qfv, lb + i * FD, Obuf);
    k_gemm<EPI_RESID><<<dim3(gm, DIMV / BN), 256, 0, stream>>>(
        Obuf, WoT + (size_t)i * DIMV * FD, x, ROWS, DIMV, FD,
        nullptr, x, nullptr);
    k_rmsnorm<<<ROWS, 256, 0, stream>>>(x, xn);
    k_gemm<EPI_BIAS_GELU_BF16><<<dim3(gm, MLPD / BN), 256, 0, stream>>>(
        xn, W1T + (size_t)i * MLPD * DIMV, hbuf, ROWS, MLPD, DIMV,
        b1 + (size_t)i * MLPD, nullptr, nullptr);
    k_gemm<EPI_BIAS_RESID><<<dim3(gm, DIMV / BN), 256, 0, stream>>>(
        hbuf, W2T + (size_t)i * DIMV * MLPD, x, ROWS, DIMV, MLPD,
        b2 + (size_t)i * DIMV, x, nullptr);
  }

  k_final<<<BATCH, 256, 0, stream>>>(x, lnh_g, lnh_b, hx);
  k_head<<<BATCH, 256, 0, stream>>>(hx, Wh, bh, (float*)d_out);
}
